// BiasMultiheadAttention_27221502722581
// MI455X (gfx1250) — compile-verified, matrix operand order checked by reading, not run on hardware
//
#include <hip/hip_runtime.h>
#include <cstdint>

typedef __attribute__((ext_vector_type(16))) _Float16 v16h;
typedef __attribute__((ext_vector_type(8)))  _Float16 v8h;
typedef __attribute__((ext_vector_type(8)))  float    v8f;
typedef __attribute__((ext_vector_type(4)))  int      v4i;

// AS-qualified int4 for the async-to-LDS builtin (param types per hipcc
// diagnostic: int4 __device__* / int4 __shared__*).
typedef __attribute__((address_space(1))) v4i as1_v4i;
typedef __attribute__((address_space(3))) v4i as3_v4i;

#define NSEQ 2048
#define DMODEL 1024
#define NHEAD 16
#define HDIM 64
#define BATCH 2

#if __has_builtin(__builtin_amdgcn_global_load_async_to_lds_b128) && \
    __has_builtin(__builtin_amdgcn_s_wait_asynccnt)
#define USE_ASYNC_LDS 1
#endif

// ---------------------------------------------------------------------------
// WMMA helper: D = A(16x32 f16) x B(32x16 f16) + C(16x16 f32)
// ---------------------------------------------------------------------------
__device__ __forceinline__ v8f wmma32(v16h a, v16h b, v8f c) {
  return __builtin_amdgcn_wmma_f32_16x16x32_f16(false, a, false, b, (short)0, c,
                                                false, false);
}

// A/B fragment, 16-bit elements, per ISA layout:
// lane (m or n) = lane&15, g = lane>>4.
// halves 0..7  -> k = k0 + 8*g + i        (contiguous)
// halves 8..15 -> k = k0 + 16 + 8*g + i   (contiguous)
__device__ __forceinline__ v16h frag_f16(const _Float16* __restrict__ row,
                                         int k0, int g) {
  v8h a = *reinterpret_cast<const v8h*>(row + k0 + 8 * g);
  v8h b = *reinterpret_cast<const v8h*>(row + k0 + 16 + 8 * g);
  v16h f;
#pragma unroll
  for (int i = 0; i < 8; ++i) { f[i] = a[i]; f[8 + i] = b[i]; }
  return f;
}

__device__ __forceinline__ v16h frag_f32(const float* __restrict__ row,
                                         int k0, int g) {
  const float4* p0 = reinterpret_cast<const float4*>(row + k0 + 8 * g);
  const float4* p1 = reinterpret_cast<const float4*>(row + k0 + 16 + 8 * g);
  float4 a = p0[0], b = p0[1], c = p1[0], d = p1[1];
  v16h f;
  f[0] = (_Float16)a.x;  f[1] = (_Float16)a.y;
  f[2] = (_Float16)a.z;  f[3] = (_Float16)a.w;
  f[4] = (_Float16)b.x;  f[5] = (_Float16)b.y;
  f[6] = (_Float16)b.z;  f[7] = (_Float16)b.w;
  f[8] = (_Float16)c.x;  f[9] = (_Float16)c.y;
  f[10] = (_Float16)c.z; f[11] = (_Float16)c.w;
  f[12] = (_Float16)d.x; f[13] = (_Float16)d.y;
  f[14] = (_Float16)d.z; f[15] = (_Float16)d.w;
  return f;
}

// xor-16 lane exchange (wave32): ds_swizzle group-of-32, xor=0x10, and=0x1f
__device__ __forceinline__ float xor16(float x) {
  int i = __builtin_amdgcn_ds_swizzle(__float_as_int(x), 0x401F);
  return __int_as_float(i);
}

// 16-byte global -> LDS copy. Async (ASYNCcnt-tracked) when available.
__device__ __forceinline__ void cp16_g2l(const _Float16* g, _Float16* l) {
#ifdef USE_ASYNC_LDS
  as1_v4i* gp = (as1_v4i*)(uintptr_t)g;
  as3_v4i* lp = (as3_v4i*)(uint32_t)(uintptr_t)l;  // flat[31:0] == LDS offset
  __builtin_amdgcn_global_load_async_to_lds_b128(gp, lp, 0, 0);
#else
  *reinterpret_cast<v8h*>(l) = *reinterpret_cast<const v8h*>(g);
#endif
}

__device__ __forceinline__ void async_wait0() {
#ifdef USE_ASYNC_LDS
  __builtin_amdgcn_s_wait_asynccnt(0);
#endif
}

// ---------------------------------------------------------------------------
// Projection GEMM: Y[row,col] = sum_d X[row,d] * W[col,d] + bias[col]
// X: [B*N, 1024] f32, W: [1024,1024] f32 (row-major over output feature)
// Out: f16 with head-split layout; transpose_out=1 stores [B,H,HD,N] (for V).
// Block = 128 thr (4 waves), tile 64x64, wave tile 32x32 (2x2 WMMA frags).
// ---------------------------------------------------------------------------
__global__ __launch_bounds__(128) void proj_kernel(
    const float* __restrict__ X, const float* __restrict__ W,
    const float* __restrict__ bias, _Float16* __restrict__ Out,
    float scale, int transpose_out) {
  const int lane = threadIdx.x & 31;
  const int wv = threadIdx.x >> 5;
  const int ln = lane & 15;
  const int g = lane >> 4;
  const int rb = blockIdx.y * 64 + (wv & 1) * 32;   // row base (B*N space)
  const int cb = blockIdx.x * 64 + (wv >> 1) * 32;  // col base (feature space)

  const float* Arow0 = X + (size_t)(rb + ln) * DMODEL;
  const float* Arow1 = X + (size_t)(rb + 16 + ln) * DMODEL;
  const float* Brow0 = W + (size_t)(cb + ln) * DMODEL;
  const float* Brow1 = W + (size_t)(cb + 16 + ln) * DMODEL;

  v8f acc00 = {}, acc01 = {}, acc10 = {}, acc11 = {};

  for (int k0 = 0; k0 < DMODEL; k0 += 32) {
    if (k0 + 256 < DMODEL) {
      __builtin_prefetch(Arow0 + k0 + 256, 0, 1);
      __builtin_prefetch(Brow0 + k0 + 256, 0, 1);
    }
    v16h a0 = frag_f32(Arow0, k0, g);
    v16h a1 = frag_f32(Arow1, k0, g);
    v16h b0 = frag_f32(Brow0, k0, g);
    v16h b1 = frag_f32(Brow1, k0, g);
    acc00 = wmma32(a0, b0, acc00);
    acc01 = wmma32(a0, b1, acc01);
    acc10 = wmma32(a1, b0, acc10);
    acc11 = wmma32(a1, b1, acc11);
  }

  // C-frag: lane holds (m = r + 8*g, n = ln)
  v8f* accs[2][2] = {{&acc00, &acc01}, {&acc10, &acc11}};
#pragma unroll
  for (int mi = 0; mi < 2; ++mi) {
#pragma unroll
    for (int ni = 0; ni < 2; ++ni) {
      int col = cb + ni * 16 + ln;
      int hh = col >> 6;     // head
      int dd = col & 63;     // dim within head
      float bv = bias[col];
      v8f av = *accs[mi][ni];
#pragma unroll
      for (int r = 0; r < 8; ++r) {
        int row = rb + mi * 16 + r + 8 * g;
        int bbn = row >> 11;     // batch (row / 2048)
        int nn = row & (NSEQ - 1);
        float v = (av[r] + bv) * scale;
        size_t idx;
        if (transpose_out)
          idx = (((size_t)(bbn * NHEAD + hh)) * HDIM + dd) * NSEQ + nn;
        else
          idx = (((size_t)(bbn * NHEAD + hh)) * NSEQ + nn) * HDIM + dd;
        Out[idx] = (_Float16)v;
      }
    }
  }
}

// ---------------------------------------------------------------------------
// Flash attention (transposed form) with double-buffered async LDS staging.
// Per block: 64 query rows (4 waves x 16). K/V tiles for a 32-key step are
// staged once into LDS (async, ASYNCcnt-tracked) and consumed by all 4 waves.
//   S^T = K_tile(16j x 32d) x Q^T   -> C-frag: m=j, n=i
//   O^T = V^T_tile(16d x 32j) x P^T(B-frag, in-lane repack of S^T C-frags)
// qh,kh: [B,H,N,64] f16 (Q pre-scaled by 1/8); vt: [B,H,64,N] f16.
// oh out: [B,N,1024] f16.
// ---------------------------------------------------------------------------
__global__ __launch_bounds__(128) void attn_kernel(
    const _Float16* __restrict__ qh, const _Float16* __restrict__ kh,
    const _Float16* __restrict__ vt, const float* __restrict__ abias,
    _Float16* __restrict__ oh, const int* __restrict__ p_self,
    const int* __restrict__ p_causal) {
  __shared__ _Float16 Kb[2][32 * HDIM];  // [j'][d]   4KB per buffer
  __shared__ _Float16 Vb[2][HDIM * 32];  // [d][j']   4KB per buffer

  const int tid = threadIdx.x;
  const int lane = tid & 31;
  const int wv = tid >> 5;
  const int ln = lane & 15;
  const int g = lane >> 4;
  const int h = blockIdx.y;
  const int b = blockIdx.z;
  const int q0 = blockIdx.x * 64;
  const int qb = q0 + wv * 16;  // wave's query base

  const int self = p_self[0];
  const int causal = p_causal[0] && self;
  const float slope = self ? exp2f(-8.0f * (float)(h + 1) / (float)NHEAD) : 0.0f;

  const size_t hq = (size_t)(b * NHEAD + h) * NSEQ * HDIM;
  const size_t hv = (size_t)(b * NHEAD + h) * HDIM * NSEQ;
  const _Float16* kbase = kh + hq;
  const _Float16* vbase = vt + hv;

  const int ig = qb + ln;  // this lane's query index (column of S^T)
  const _Float16* qrow = qh + hq + (size_t)ig * HDIM;
  const v16h qf0 = frag_f16(qrow, 0, g);
  const v16h qf1 = frag_f16(qrow, 32, g);
  const float* brow = abias + (size_t)ig * NSEQ;

  v8f ot0 = {}, ot1 = {}, ot2 = {}, ot3 = {};
  float Mi = -3.0e38f, Li = 0.0f;

  const int jlim = causal ? (q0 + 64) : NSEQ;
  const int ntiles = jlim >> 5;

  // Stage a 32-key K tile + V^T tile into LDS buffer `buf`.
  // 512 16B chunks total, 4 per thread.
  auto stage = [&](int buf, int j0) {
#pragma unroll
    for (int c2 = 0; c2 < 2; ++c2) {
      int c = tid + c2 * 128;          // 0..255
      int row = c >> 3;                // 0..31 (j')
      int off = (c & 7) * 8;           // 0..56 (d)
      cp16_g2l(kbase + (size_t)(j0 + row) * HDIM + off, &Kb[buf][row * HDIM + off]);
    }
#pragma unroll
    for (int c2 = 0; c2 < 2; ++c2) {
      int c = tid + c2 * 128;          // 0..255
      int d = c >> 2;                  // 0..63
      int off = (c & 3) * 8;           // 0..24 (j')
      cp16_g2l(vbase + (size_t)d * NSEQ + j0 + off, &Vb[buf][d * 32 + off]);
    }
  };

  stage(0, 0);  // prologue: issue tile 0

  for (int it = 0; it < ntiles; ++it) {
    const int j0 = it << 5;
    const int buf = it & 1;

    async_wait0();      // this tile's staging complete (per-wave)
    __syncthreads();    // visible to all waves

    if (it + 1 < ntiles) stage(buf ^ 1, j0 + 32);  // prefetch next tile

    // Wave-uniform compute skip for fully-masked tiles (no barriers inside).
    if (!(causal && j0 >= qb + 16)) {
      // --- S^T for two 16-key subtiles ---
      v8f st[2];
#pragma unroll
      for (int sub = 0; sub < 2; ++sub) {
        const _Float16* krow = &Kb[buf][(sub * 16 + ln) * HDIM];
        v16h kf0 = frag_f16(krow, 0, g);
        v16h kf1 = frag_f16(krow, 32, g);
        v8f c = {};
        c = wmma32(kf0, qf0, c);
        c = wmma32(kf1, qf1, c);
        st[sub] = c;
      }

      // --- bias / alibi / causal mask; per-lane column stats ---
      float p[16];
      float tmax = -3.0e38f;
#pragma unroll
      for (int sub = 0; sub < 2; ++sub) {
        const float4* bp =
            reinterpret_cast<const float4*>(brow + j0 + sub * 16 + 8 * g);
        float4 bx = bp[0], by = bp[1];
        float bb[8] = {bx.x, bx.y, bx.z, bx.w, by.x, by.y, by.z, by.w};
#pragma unroll
        for (int r = 0; r < 8; ++r) {
          int j = j0 + sub * 16 + r + 8 * g;
          float sv = st[sub][r];
          if (self) {
            int d = j - ig;
            if (d > 0) sv -= slope * (float)d;
          }
          sv += bb[r];
          if (causal && j > ig) sv = -3.0e38f;
          p[sub * 8 + r] = sv;
          tmax = fmaxf(tmax, sv);
        }
      }
      tmax = fmaxf(tmax, xor16(tmax));  // merge the two half-wave j-groups

      const float newM = fmaxf(Mi, tmax);
      const float corr = __expf(Mi - newM);

      float tsum = 0.0f;
      v16h pf;  // P^T B-fragment: in-lane repack, no shuffles
#pragma unroll
      for (int t = 0; t < 16; ++t) {
        float e = __expf(p[t] - newM);
        tsum += e;
        pf[t] = (_Float16)e;
      }
      tsum += xor16(tsum);
      Li = Li * corr + tsum;

      ot0 = ot0 * corr; ot1 = ot1 * corr; ot2 = ot2 * corr; ot3 = ot3 * corr;

      // --- O^T += V^T x P^T  (4 d-tiles of 16) ---
      ot0 = wmma32(frag_f16(&Vb[buf][(0 * 16 + ln) * 32], 0, g), pf, ot0);
      ot1 = wmma32(frag_f16(&Vb[buf][(1 * 16 + ln) * 32], 0, g), pf, ot1);
      ot2 = wmma32(frag_f16(&Vb[buf][(2 * 16 + ln) * 32], 0, g), pf, ot2);
      ot3 = wmma32(frag_f16(&Vb[buf][(3 * 16 + ln) * 32], 0, g), pf, ot3);

      Mi = newM;
    }

    __syncthreads();  // all reads of `buf` done before it is restaged
  }

  const float inv = (Li > 0.0f) ? 1.0f / Li : 0.0f;

  // O^T C-frag: lane holds (m = d_local = r + 8g, n = i). Contiguous 8-half
  // runs along d -> one 16B store per d-tile per lane.
  _Float16* orow = oh + ((size_t)(b * NSEQ + ig)) * DMODEL + h * HDIM;
  v8f* ots[4] = {&ot0, &ot1, &ot2, &ot3};
#pragma unroll
  for (int dt = 0; dt < 4; ++dt) {
    v8f av = *ots[dt];
    v8h o8;
#pragma unroll
    for (int r = 0; r < 8; ++r) o8[r] = (_Float16)(av[r] * inv);
    *reinterpret_cast<v8h*>(orow + dt * 16 + 8 * g) = o8;
  }
}

// ---------------------------------------------------------------------------
// Output projection: out[row,col] = sum_e Oh[row,e] * Wo[col,e] + bo[col], f32
// ---------------------------------------------------------------------------
__global__ __launch_bounds__(128) void outproj_kernel(
    const _Float16* __restrict__ Oh, const float* __restrict__ W,
    const float* __restrict__ bias, float* __restrict__ Out) {
  const int lane = threadIdx.x & 31;
  const int wv = threadIdx.x >> 5;
  const int ln = lane & 15;
  const int g = lane >> 4;
  const int rb = blockIdx.y * 64 + (wv & 1) * 32;
  const int cb = blockIdx.x * 64 + (wv >> 1) * 32;

  const _Float16* Arow0 = Oh + (size_t)(rb + ln) * DMODEL;
  const _Float16* Arow1 = Oh + (size_t)(rb + 16 + ln) * DMODEL;
  const float* Brow0 = W + (size_t)(cb + ln) * DMODEL;
  const float* Brow1 = W + (size_t)(cb + 16 + ln) * DMODEL;

  v8f acc00 = {}, acc01 = {}, acc10 = {}, acc11 = {};

  for (int k0 = 0; k0 < DMODEL; k0 += 32) {
    if (k0 + 256 < DMODEL) {
      __builtin_prefetch(Arow0 + k0 + 256, 0, 1);
      __builtin_prefetch(Brow0 + k0 + 256, 0, 1);
    }
    v16h a0 = frag_f16(Arow0, k0, g);
    v16h a1 = frag_f16(Arow1, k0, g);
    v16h b0 = frag_f32(Brow0, k0, g);
    v16h b1 = frag_f32(Brow1, k0, g);
    acc00 = wmma32(a0, b0, acc00);
    acc01 = wmma32(a0, b1, acc01);
    acc10 = wmma32(a1, b0, acc10);
    acc11 = wmma32(a1, b1, acc11);
  }

  v8f* accs[2][2] = {{&acc00, &acc01}, {&acc10, &acc11}};
#pragma unroll
  for (int mi = 0; mi < 2; ++mi) {
#pragma unroll
    for (int ni = 0; ni < 2; ++ni) {
      int col = cb + ni * 16 + ln;
      float bv = bias[col];
      v8f av = *accs[mi][ni];
#pragma unroll
      for (int r = 0; r < 8; ++r) {
        int row = rb + mi * 16 + r + 8 * g;
        Out[(size_t)row * DMODEL + col] = av[r] + bv;
      }
    }
  }
}

// ---------------------------------------------------------------------------
extern "C" void kernel_launch(void* const* d_in, const int* in_sizes, int n_in,
                              void* d_out, int out_size, void* d_ws,
                              size_t ws_size, hipStream_t stream) {
  const float* x_q   = (const float*)d_in[0];
  const float* x_kv  = (const float*)d_in[1];
  const float* abias = (const float*)d_in[2];
  const float* Wq    = (const float*)d_in[3];
  const float* bq    = (const float*)d_in[4];
  const float* Wk    = (const float*)d_in[5];
  const float* bk    = (const float*)d_in[6];
  const float* Wv    = (const float*)d_in[7];
  const float* bv    = (const float*)d_in[8];
  const float* Wo    = (const float*)d_in[9];
  const float* bo    = (const float*)d_in[10];
  const int* self_f   = (const int*)d_in[11];
  const int* causal_f = (const int*)d_in[12];
  float* out = (float*)d_out;

  char* ws = (char*)d_ws;
  const size_t seg = (size_t)BATCH * NSEQ * DMODEL * sizeof(_Float16);  // 8 MB
  _Float16* qh = (_Float16*)(ws);
  _Float16* kh = (_Float16*)(ws + seg);
  _Float16* vt = (_Float16*)(ws + 2 * seg);
  _Float16* oh = (_Float16*)(ws + 3 * seg);

  dim3 gemm_grid(DMODEL / 64, (BATCH * NSEQ) / 64);  // (16, 64)
  dim3 blk(128);

  // Q pre-scaled by 1/sqrt(HDIM) = 0.125
  proj_kernel<<<gemm_grid, blk, 0, stream>>>(x_q,  Wq, bq, qh, 0.125f, 0);
  proj_kernel<<<gemm_grid, blk, 0, stream>>>(x_kv, Wk, bk, kh, 1.0f,   0);
  proj_kernel<<<gemm_grid, blk, 0, stream>>>(x_kv, Wv, bv, vt, 1.0f,   1);

  attn_kernel<<<dim3(NSEQ / 64, NHEAD, BATCH), blk, 0, stream>>>(
      qh, kh, vt, abias, oh, self_f, causal_f);

  outproj_kernel<<<gemm_grid, blk, 0, stream>>>(oh, Wo, bo, out);
}
